// MAGClassifier_18743237279744
// MI455X (gfx1250) — matrix operand order repaired and checked by reading, not compiled, not run on hardware
//
#include <hip/hip_runtime.h>
#include <hip/hip_bf16.h>

typedef _Float16 half_t;
typedef __attribute__((ext_vector_type(16))) _Float16 v16h;
typedef __attribute__((ext_vector_type(8)))  _Float16 v8h;
typedef __attribute__((ext_vector_type(8)))  float    v8f;

#define WMMA_F16(a, b, c) \
  __builtin_amdgcn_wmma_f32_16x16x32_f16(false, (a), false, (b), (short)0, (c), false, false)

// ---------------------------------------------------------------------------
// WMMA fragment loaders (wave32 layouts per CDNA5 ISA 7.12.2), vectorized.
// A 16x32 f16: lane L -> row L&15; elements 0..7  = K (L>>4)*8 .. +7 (contiguous)
//                                  elements 8..15 = K 16+(L>>4)*8 .. +7 (contiguous)
// B 32x16 f16: lane L -> col L&15; elements 0..7 = K (L>>4)*8 .. +7, elements 8..15 =
//              K 16+(L>>4)*8 .. +7 (KxN matrix stored transposed as [N][K])
// C/D 16x16 f32: lane L -> col L&15; VGPR v -> row (L>>4)*8 + v
// ---------------------------------------------------------------------------
__device__ inline v16h cat8(v8h lo, v8h hi) {
  return __builtin_shufflevector(lo, hi, 0, 1, 2, 3, 4, 5, 6, 7,
                                 8, 9, 10, 11, 12, 13, 14, 15);
}

__device__ inline v16h frag_a(const half_t* base, int ld) {
  int lane = threadIdx.x & 31, row = lane & 15, hi = lane >> 4;
  const half_t* p = base + row * ld + hi * 8;
  v8h lo = *(const v8h*)(p);
  v8h hh = *(const v8h*)(p + 16);
  return cat8(lo, hh);
}

// A 16x32 where only K<16 is valid (rest zero) -- dh=16 attention tiles.
__device__ inline v16h frag_a_pad16(const half_t* base, int ld) {
  int lane = threadIdx.x & 31, row = lane & 15, hi = lane >> 4;
  v8h lo = *(const v8h*)(base + row * ld + hi * 8);
  v8h zz = (_Float16)0.0f;
  return cat8(lo, zz);
}

// B 32x16 where memory holds the KxN matrix transposed: baseT[n*ldT + k].
__device__ inline v16h frag_bT(const half_t* baseT, int ldT) {
  int lane = threadIdx.x & 31, col = lane & 15, hi = lane >> 4;
  const half_t* p = baseT + col * ldT + hi * 8;
  v8h lo = *(const v8h*)(p);
  v8h hh = *(const v8h*)(p + 16);
  return cat8(lo, hh);
}

// B 32x16 with only K<16 valid (zeros above): one 16B load per lane (K (L>>4)*8 .. +7).
__device__ inline v16h frag_bT_pad16(const half_t* baseT, int ldT) {
  int lane = threadIdx.x & 31, col = lane & 15, hi = lane >> 4;
  v8h lo = *(const v8h*)(baseT + col * ldT + hi * 8);
  v8h zz = (_Float16)0.0f;
  return cat8(lo, zz);
}

// ---------------------------------------------------------------------------
// Convert f32 [K x N] weight to f16 transposed [N x K].
__global__ void cvt_f16_t(const float* __restrict__ s, half_t* __restrict__ d,
                          int K, int N) {
  int i = blockIdx.x * 256 + threadIdx.x;
  if (i < K * N) {
    int k = i / N, n = i - k * N;
    d[n * K + k] = (half_t)s[i];
  }
}

// Edge feature gather + Linear(160->128)+ReLU + Linear(128->128). 16 edges/block.
__global__ __launch_bounds__(32) void edge_mlp(
    const float* __restrict__ x, const int* __restrict__ ei, const float* __restrict__ ea,
    const half_t* __restrict__ W1T, const float* __restrict__ b1,
    const half_t* __restrict__ W2T, const float* __restrict__ b2,
    float* __restrict__ Yf, half_t* __restrict__ Yh) {
  __shared__ half_t feat[16 * 160];
  __shared__ half_t hid[16 * 128];
  int rowbase = blockIdx.x * 16;
  int tid = threadIdx.x;
  for (int i = tid; i < 16 * 160; i += 32) {
    int r = i / 160, k = i - r * 160, e = rowbase + r;
    float v;
    if (k < 64)       v = x[ei[e] * 64 + k];
    else if (k < 128) v = x[ei[8192 + e] * 64 + (k - 64)];
    else              v = ea[e * 32 + (k - 128)];
    feat[r * 160 + k] = (half_t)v;
  }
  __syncthreads();
  int lane = tid & 31, col = lane & 15, hi = lane >> 4;
  v16h a1[5];
#pragma unroll
  for (int kt = 0; kt < 5; ++kt) a1[kt] = frag_a(feat + kt * 32, 160);
#pragma unroll
  for (int nt = 0; nt < 8; ++nt) {
    v8f acc = {};
#pragma unroll
    for (int kt = 0; kt < 5; ++kt) {
      v16h b = frag_bT(W1T + (nt * 16) * 160 + kt * 32, 160);
      acc = WMMA_F16(a1[kt], b, acc);
    }
    float bb = b1[nt * 16 + col];
#pragma unroll
    for (int v = 0; v < 8; ++v)
      hid[(hi * 8 + v) * 128 + nt * 16 + col] = (half_t)fmaxf(acc[v] + bb, 0.0f);
  }
  __syncthreads();
  v16h a2[4];
#pragma unroll
  for (int kt = 0; kt < 4; ++kt) a2[kt] = frag_a(hid + kt * 32, 128);
#pragma unroll
  for (int nt = 0; nt < 8; ++nt) {
    v8f acc = {};
#pragma unroll
    for (int kt = 0; kt < 4; ++kt) {
      v16h b = frag_bT(W2T + (nt * 16) * 128 + kt * 32, 128);
      acc = WMMA_F16(a2[kt], b, acc);
    }
    float bb = b2[nt * 16 + col];
#pragma unroll
    for (int v = 0; v < 8; ++v) {
      int idx = (rowbase + hi * 8 + v) * 128 + nt * 16 + col;
      float o = acc[v] + bb;
      Yf[idx] = o;
      Yh[idx] = (half_t)o;
    }
  }
}

// Fused Q/K/V projection: A-frags of X loaded once, 96 wmmas per wave.
// Q,K row-major f16; V stored transposed Vt[n*8192 + row] for the P@V B-frags.
__global__ __launch_bounds__(256) void qkv_proj(
    const half_t* __restrict__ X,
    const half_t* __restrict__ WqT, const half_t* __restrict__ WkT,
    const half_t* __restrict__ WvT,
    const float* __restrict__ bq, const float* __restrict__ bk,
    const float* __restrict__ bv,
    half_t* __restrict__ Q, half_t* __restrict__ Ko, half_t* __restrict__ Vt) {
  int wave = threadIdx.x >> 5;
  int rowbase = blockIdx.x * 128 + wave * 16;
  int lane = threadIdx.x & 31, col = lane & 15, hi = lane >> 4;
  v16h a[4];
#pragma unroll
  for (int kt = 0; kt < 4; ++kt) a[kt] = frag_a(X + rowbase * 128 + kt * 32, 128);
#pragma unroll
  for (int nt = 0; nt < 8; ++nt) {
    v8f acc = {};
#pragma unroll
    for (int kt = 0; kt < 4; ++kt)
      acc = WMMA_F16(a[kt], frag_bT(WqT + (nt * 16) * 128 + kt * 32, 128), acc);
    float bb = bq[nt * 16 + col];
#pragma unroll
    for (int v = 0; v < 8; ++v)
      Q[(rowbase + hi * 8 + v) * 128 + nt * 16 + col] = (half_t)(acc[v] + bb);
  }
#pragma unroll
  for (int nt = 0; nt < 8; ++nt) {
    v8f acc = {};
#pragma unroll
    for (int kt = 0; kt < 4; ++kt)
      acc = WMMA_F16(a[kt], frag_bT(WkT + (nt * 16) * 128 + kt * 32, 128), acc);
    float bb = bk[nt * 16 + col];
#pragma unroll
    for (int v = 0; v < 8; ++v)
      Ko[(rowbase + hi * 8 + v) * 128 + nt * 16 + col] = (half_t)(acc[v] + bb);
  }
#pragma unroll
  for (int nt = 0; nt < 8; ++nt) {
    v8f acc = {};
#pragma unroll
    for (int kt = 0; kt < 4; ++kt)
      acc = WMMA_F16(a[kt], frag_bT(WvT + (nt * 16) * 128 + kt * 32, 128), acc);
    float bb = bv[nt * 16 + col];
    v8h out;
#pragma unroll
    for (int v = 0; v < 8; ++v) out[v] = (half_t)(acc[v] + bb);
    *(v8h*)(Vt + (nt * 16 + col) * 8192 + rowbase + hi * 8) = out;
  }
}

// Y_f16 = X@W + bias, row-major out. 8 waves/block, 1 row-tile per wave.
__global__ __launch_bounds__(256) void gemm128(
    const half_t* __restrict__ X, const half_t* __restrict__ WT,
    const float* __restrict__ bias, half_t* __restrict__ Y) {
  int wave = threadIdx.x >> 5;
  int rowbase = blockIdx.x * 128 + wave * 16;
  int lane = threadIdx.x & 31, col = lane & 15, hi = lane >> 4;
  v16h a[4];
#pragma unroll
  for (int kt = 0; kt < 4; ++kt) a[kt] = frag_a(X + rowbase * 128 + kt * 32, 128);
#pragma unroll
  for (int nt = 0; nt < 8; ++nt) {
    v8f acc = {};
#pragma unroll
    for (int kt = 0; kt < 4; ++kt) {
      v16h b = frag_bT(WT + (nt * 16) * 128 + kt * 32, 128);
      acc = WMMA_F16(a[kt], b, acc);
    }
    float bb = bias[nt * 16 + col];
#pragma unroll
    for (int v = 0; v < 8; ++v)
      Y[(rowbase + hi * 8 + v) * 128 + nt * 16 + col] = (half_t)(acc[v] + bb);
  }
}

// Same GEMM but stores the result transposed: Yt[n*8192 + row] (one v8h store/tile).
__global__ __launch_bounds__(256) void gemm128_t(
    const half_t* __restrict__ X, const half_t* __restrict__ WT,
    const float* __restrict__ bias, half_t* __restrict__ Yt) {
  int wave = threadIdx.x >> 5;
  int rowbase = blockIdx.x * 128 + wave * 16;
  int lane = threadIdx.x & 31, col = lane & 15, hi = lane >> 4;
  v16h a[4];
#pragma unroll
  for (int kt = 0; kt < 4; ++kt) a[kt] = frag_a(X + rowbase * 128 + kt * 32, 128);
#pragma unroll
  for (int nt = 0; nt < 8; ++nt) {
    v8f acc = {};
#pragma unroll
    for (int kt = 0; kt < 4; ++kt) {
      v16h b = frag_bT(WT + (nt * 16) * 128 + kt * 32, 128);
      acc = WMMA_F16(a[kt], b, acc);
    }
    float bb = bias[nt * 16 + col];
    v8h out;
#pragma unroll
    for (int v = 0; v < 8; ++v) out[v] = (half_t)(acc[v] + bb);
    *(v8h*)(Yt + (nt * 16 + col) * 8192 + rowbase + hi * 8) = out;
  }
}

// Y = LN(res + maybe_relu(X@W + bias)) * gamma + beta ; dual f32/f16 store.
__global__ __launch_bounds__(256) void gemm_res_ln(
    const half_t* __restrict__ X, const half_t* __restrict__ WT,
    const float* __restrict__ bias, const float* __restrict__ res,
    const float* __restrict__ gamma, const float* __restrict__ beta,
    float* __restrict__ Yf, half_t* __restrict__ Yh, int doRelu) {
  int wave = threadIdx.x >> 5;
  int rowbase = blockIdx.x * 128 + wave * 16;
  int lane = threadIdx.x & 31, col = lane & 15, hi = lane >> 4;
  v8f acc[8];
#pragma unroll
  for (int nt = 0; nt < 8; ++nt) acc[nt] = v8f{};
#pragma unroll
  for (int kt = 0; kt < 4; ++kt) {
    v16h a = frag_a(X + rowbase * 128 + kt * 32, 128);
#pragma unroll
    for (int nt = 0; nt < 8; ++nt) {
      v16h b = frag_bT(WT + (nt * 16) * 128 + kt * 32, 128);
      acc[nt] = WMMA_F16(a, b, acc[nt]);
    }
  }
#pragma unroll
  for (int nt = 0; nt < 8; ++nt) {
    float bb = bias[nt * 16 + col];
#pragma unroll
    for (int v = 0; v < 8; ++v) {
      float t = acc[nt][v] + bb;
      if (doRelu) t = fmaxf(t, 0.0f);
      t += res[(rowbase + hi * 8 + v) * 128 + nt * 16 + col];
      acc[nt][v] = t;
    }
  }
  float mean[8], rstd[8];
#pragma unroll
  for (int v = 0; v < 8; ++v) {
    float s = 0.0f;
#pragma unroll
    for (int nt = 0; nt < 8; ++nt) s += acc[nt][v];
    s += __shfl_xor(s, 1); s += __shfl_xor(s, 2);
    s += __shfl_xor(s, 4); s += __shfl_xor(s, 8);
    mean[v] = s * (1.0f / 128.0f);
  }
#pragma unroll
  for (int v = 0; v < 8; ++v) {
    float q = 0.0f;
#pragma unroll
    for (int nt = 0; nt < 8; ++nt) { float d = acc[nt][v] - mean[v]; q += d * d; }
    q += __shfl_xor(q, 1); q += __shfl_xor(q, 2);
    q += __shfl_xor(q, 4); q += __shfl_xor(q, 8);
    rstd[v] = rsqrtf(q * (1.0f / 128.0f) + 1e-5f);
  }
#pragma unroll
  for (int nt = 0; nt < 8; ++nt) {
    float gg = gamma[nt * 16 + col], be = beta[nt * 16 + col];
#pragma unroll
    for (int v = 0; v < 8; ++v) {
      float o = (acc[nt][v] - mean[v]) * rstd[v] * gg + be;
      int idx = (rowbase + hi * 8 + v) * 128 + nt * 16 + col;
      Yf[idx] = o;
      Yh[idx] = (half_t)o;
    }
  }
}

// Flash-attention; 8 waves/block = 8 query tiles of the same (graph, head).
// V is pre-transposed (Vt[dh_col * 8192 + edge]) so the P@V B-frag is two 16B loads per lane.
__global__ __launch_bounds__(256) void attn_kernel(
    const half_t* __restrict__ Q, const half_t* __restrict__ K,
    const half_t* __restrict__ Vt, const int* __restrict__ ei,
    half_t* __restrict__ O, int use_mask) {
  int wave = threadIdx.x >> 5;
  int qt = blockIdx.x * 8 + wave;
  int hd = blockIdx.y, g = blockIdx.z;
  int qbase = g * 1024 + qt * 16;
  int lane = threadIdx.x & 31, col = lane & 15, hi = lane >> 4;
  __shared__ int ks_s[1024], kd_s[1024];
  __shared__ half_t pS[8][16 * 32];
  if (use_mask) {
    for (int i = threadIdx.x; i < 1024; i += 256) {
      ks_s[i] = ei[g * 1024 + i];
      kd_s[i] = ei[8192 + g * 1024 + i];
    }
  }
  __syncthreads();
  half_t* myP = pS[wave];
  v16h aq = frag_a_pad16(Q + qbase * 128 + hd * 16, 128);
  int qs[8], qd[8];
#pragma unroll
  for (int v = 0; v < 8; ++v) {
    int e = qbase + hi * 8 + v;
    qs[v] = ei[e];
    qd[v] = ei[8192 + e];
  }
  v8f accO = {};
  float m[8], l[8];
#pragma unroll
  for (int v = 0; v < 8; ++v) { m[v] = -1e30f; l[v] = 0.0f; }

  for (int kc = 0; kc < 32; ++kc) {
    int kbase = g * 1024 + kc * 32;
    if (kc + 1 < 32) {
      // warm WGP$/L2 for the next key chunk (global_prefetch_b8)
      __builtin_prefetch(K + (kbase + 32 + col) * 128 + hd * 16, 0, 3);
      __builtin_prefetch(K + (kbase + 48 + col) * 128 + hd * 16, 0, 3);
      __builtin_prefetch(Vt + (hd * 16 + col) * 8192 + kbase + 32, 0, 3);
    }
    float s2[2][8];
#pragma unroll
    for (int t = 0; t < 2; ++t) {
      v16h bK = frag_bT_pad16(K + (kbase + t * 16) * 128 + hd * 16, 128);
      v8f z = {};
      v8f s = WMMA_F16(aq, bK, z);
      int kloc = kc * 32 + t * 16 + col;
      int ksv = 0, kdv = 0;
      if (use_mask) { ksv = ks_s[kloc]; kdv = kd_s[kloc]; }
#pragma unroll
      for (int v = 0; v < 8; ++v) {
        float sc = s[v] * 0.25f;  // 1/sqrt(dh=16)
        if (use_mask) {
          bool adj = (qs[v] == ksv) | (qs[v] == kdv) | (qd[v] == ksv) | (qd[v] == kdv);
          sc = adj ? sc : -1e9f;
        }
        s2[t][v] = sc;
      }
    }
#pragma unroll
    for (int v = 0; v < 8; ++v) {
      float cm = fmaxf(s2[0][v], s2[1][v]);
      cm = fmaxf(cm, __shfl_xor(cm, 1)); cm = fmaxf(cm, __shfl_xor(cm, 2));
      cm = fmaxf(cm, __shfl_xor(cm, 4)); cm = fmaxf(cm, __shfl_xor(cm, 8));
      float mn = fmaxf(m[v], cm);
      float f  = __expf(m[v] - mn);
      float p0 = __expf(s2[0][v] - mn);
      float p1 = __expf(s2[1][v] - mn);
      float rs = p0 + p1;
      rs += __shfl_xor(rs, 1); rs += __shfl_xor(rs, 2);
      rs += __shfl_xor(rs, 4); rs += __shfl_xor(rs, 8);
      m[v] = mn;
      l[v] = l[v] * f + rs;
      accO[v] *= f;
      myP[(hi * 8 + v) * 32 + col]      = (half_t)p0;
      myP[(hi * 8 + v) * 32 + 16 + col] = (half_t)p1;
    }
    // pS write->read is same-wave; LDS ops are in-order within a wave.
    v16h aP = frag_a(myP, 32);
    v16h bV = frag_bT(Vt + (hd * 16) * 8192 + kbase, 8192);
    accO = WMMA_F16(aP, bV, accO);
  }
#pragma unroll
  for (int v = 0; v < 8; ++v) {
    float o = accO[v] / l[v];
    O[(qbase + hi * 8 + v) * 128 + hd * 16 + col] = (half_t)o;
  }
}

// --------------------------- scalar tail (8 rows) ---------------------------
__device__ inline float blk_sum(float v, float* buf) {
  int tid = threadIdx.x;
  __syncthreads();
  buf[tid] = v;
  __syncthreads();
  for (int st = 64; st > 0; st >>= 1) {
    if (tid < st) buf[tid] += buf[tid + st];
    __syncthreads();
  }
  float r = buf[0];
  __syncthreads();
  return r;
}

__global__ void pma_attn(const float* __restrict__ seed, const float* __restrict__ wq,
                         const float* __restrict__ bq, const half_t* __restrict__ Kp,
                         const half_t* __restrict__ Vpt, float* __restrict__ pooled) {
  int g = blockIdx.x, tid = threadIdx.x;  // 128 threads
  __shared__ float qrow[128];
  __shared__ float sc[1024];
  __shared__ float red[128];
  __shared__ float part[16][8];
  float a = bq[tid];
  for (int k = 0; k < 128; ++k) a += seed[k] * wq[k * 128 + tid];
  qrow[tid] = a;
  __syncthreads();
  for (int hd = 0; hd < 8; ++hd) {
    float lmax = -1e30f;
    for (int j = tid; j < 1024; j += 128) {
      const half_t* kp = Kp + (size_t)(g * 1024 + j) * 128 + hd * 16;
      float s = 0.0f;
      for (int d = 0; d < 16; ++d) s += qrow[hd * 16 + d] * (float)kp[d];
      s *= 0.25f;
      sc[j] = s;
      lmax = fmaxf(lmax, s);
    }
    __syncthreads();
    red[tid] = lmax;
    __syncthreads();
    for (int st = 64; st > 0; st >>= 1) {
      if (tid < st) red[tid] = fmaxf(red[tid], red[tid + st]);
      __syncthreads();
    }
    float mx = red[0];
    __syncthreads();
    float lsum = 0.0f;
    for (int j = tid; j < 1024; j += 128) {
      float p = __expf(sc[j] - mx);
      sc[j] = p;
      lsum += p;
    }
    float sm = blk_sum(lsum, red);
    int d = tid >> 3, grp = tid & 7;
    float oacc = 0.0f;
    for (int j = grp * 128; j < grp * 128 + 128; ++j)
      oacc += sc[j] * (float)Vpt[(size_t)(hd * 16 + d) * 8192 + g * 1024 + j];
    part[d][grp] = oacc;
    __syncthreads();
    if (tid < 16) {
      float s = 0.0f;
      for (int q = 0; q < 8; ++q) s += part[tid][q];
      pooled[g * 128 + hd * 16 + tid] = s / sm;
    }
    __syncthreads();
  }
}

__global__ void pma_tail(const float* __restrict__ seed, const float* __restrict__ pooled,
    const float* __restrict__ wo, const float* __restrict__ bo,
    const float* __restrict__ g1, const float* __restrict__ b1,
    const float* __restrict__ wf, const float* __restrict__ bf,
    const float* __restrict__ g2, const float* __restrict__ b2,
    const float* __restrict__ o1w, const float* __restrict__ o1b,
    const float* __restrict__ o2w, const float* __restrict__ o2b,
    float* __restrict__ out) {
  __shared__ float buf[128];
  __shared__ float rowv[128];
  __shared__ float h1[128];
  __shared__ float z2[128];
  int tid = threadIdx.x;  // 128 threads
  for (int g = 0; g < 8; ++g) {
    rowv[tid] = pooled[g * 128 + tid];
    __syncthreads();
    float a = bo[tid];
    for (int k = 0; k < 128; ++k) a += rowv[k] * wo[k * 128 + tid];
    float v = seed[tid] + a;
    float mean = blk_sum(v, buf) * (1.0f / 128.0f);
    float d = v - mean;
    float var = blk_sum(d * d, buf) * (1.0f / 128.0f);
    float hv = d * rsqrtf(var + 1e-5f) * g1[tid] + b1[tid];
    h1[tid] = hv;
    __syncthreads();
    float f = bf[tid];
    for (int k = 0; k < 128; ++k) f += h1[k] * wf[k * 128 + tid];
    float v2 = hv + fmaxf(f, 0.0f);
    float mean2 = blk_sum(v2, buf) * (1.0f / 128.0f);
    float d2 = v2 - mean2;
    float var2 = blk_sum(d2 * d2, buf) * (1.0f / 128.0f);
    float zv = d2 * rsqrtf(var2 + 1e-5f) * g2[tid] + b2[tid];
    z2[tid] = zv;
    __syncthreads();
    float o1 = o1b[tid];
    for (int k = 0; k < 128; ++k) o1 += z2[k] * o1w[k * 128 + tid];
    o1 = fmaxf(o1, 0.0f);
    float tot = blk_sum(o1 * o2w[tid], buf);
    if (tid == 0) out[g] = tot + o2b[0];
    __syncthreads();
  }
}

// ---------------------------------------------------------------------------
extern "C" void kernel_launch(void* const* d_in, const int* in_sizes, int n_in,
                              void* d_out, int out_size, void* d_ws, size_t ws_size,
                              hipStream_t stream) {
  (void)in_sizes; (void)n_in; (void)out_size;
  auto F = [&](int i) { return (const float*)d_in[i]; };
  const float* x  = F(0);
  const int*   ei = (const int*)d_in[1];
  const float* ea = F(2);
  // d_in[3] = node_batch (unused: edges are graph-sorted, g = e/1024)
  // params flat order: in1_w,in1_b,in2_w,in2_b, mab0..2{wq,wk,wv,wo,wf,bq,bk,bv,bo,bf,g1,b1,g2,b2},
  //                    pma{...}, seed, out1_w,out1_b,out2_w,out2_b

  char* base = (char*)d_ws;
  size_t off = 0;
  auto alloc = [&](size_t bytes) -> void* {
    void* p = base + off;
    off += (bytes + 255) & ~(size_t)255;
    return p;
  };
  half_t* Win1T = (half_t*)alloc(160 * 128 * 2);
  half_t* Win2T = (half_t*)alloc(128 * 128 * 2);
  half_t* Wm[3][5];
  for (int m = 0; m < 3; ++m)
    for (int j = 0; j < 5; ++j) Wm[m][j] = (half_t*)alloc(128 * 128 * 2);
  half_t* WpkT = (half_t*)alloc(128 * 128 * 2);
  half_t* WpvT = (half_t*)alloc(128 * 128 * 2);
  const size_t MH = (size_t)8192 * 128;
  float*  bufF[2]; half_t* bufH[2];
  bufF[0] = (float*)alloc(MH * 4);  bufH[0] = (half_t*)alloc(MH * 2);
  bufF[1] = (float*)alloc(MH * 4);  bufH[1] = (half_t*)alloc(MH * 2);
  float*  Tf = (float*)alloc(MH * 4);
  half_t* Th = (half_t*)alloc(MH * 2);
  half_t* Qh = (half_t*)alloc(MH * 2);
  half_t* Kh = (half_t*)alloc(MH * 2);
  half_t* Vt = (half_t*)alloc(MH * 2);
  half_t* Oh = (half_t*)alloc(MH * 2);
  float*  pooled = (float*)alloc(8 * 128 * 4);
  if (off > ws_size) return;

  auto cvtT = [&](const float* s, half_t* d, int K) {
    cvt_f16_t<<<(K * 128 + 255) / 256, 256, 0, stream>>>(s, d, K, 128);
  };
  cvtT(F(4), Win1T, 160);
  cvtT(F(6), Win2T, 128);
  for (int m = 0; m < 3; ++m) {
    int mb = 8 + m * 14;
    for (int j = 0; j < 5; ++j) cvtT(F(mb + j), Wm[m][j], 128);
  }
  cvtT(F(50 + 1), WpkT, 128);  // pma wk
  cvtT(F(50 + 2), WpvT, 128);  // pma wv

  edge_mlp<<<512, 32, 0, stream>>>(x, ei, ea, Win1T, F(5), Win2T, F(7), bufF[0], bufH[0]);

  int cur = 0;
  for (int m = 0; m < 3; ++m) {
    int mb = 8 + m * 14;
    qkv_proj<<<64, 256, 0, stream>>>(bufH[cur], Wm[m][0], Wm[m][1], Wm[m][2],
                                     F(mb + 5), F(mb + 6), F(mb + 7), Qh, Kh, Vt);
    attn_kernel<<<dim3(8, 8, 8), 256, 0, stream>>>(Qh, Kh, Vt, ei, Oh, (m < 2) ? 1 : 0);
    gemm_res_ln<<<64, 256, 0, stream>>>(Oh, Wm[m][3], F(mb + 8), bufF[cur],
                                        F(mb + 10), F(mb + 11), Tf, Th, 0);
    gemm_res_ln<<<64, 256, 0, stream>>>(Th, Wm[m][4], F(mb + 9), Tf,
                                        F(mb + 12), F(mb + 13), bufF[1 - cur], bufH[1 - cur], 1);
    cur ^= 1;
  }

  gemm128  <<<64, 256, 0, stream>>>(bufH[cur], WpkT, F(50 + 6), Kh);
  gemm128_t<<<64, 256, 0, stream>>>(bufH[cur], WpvT, F(50 + 7), Vt);
  pma_attn<<<8, 128, 0, stream>>>(F(64), F(50 + 0), F(50 + 5), Kh, Vt, pooled);
  pma_tail<<<1, 128, 0, stream>>>(F(64), pooled,
                                  F(50 + 3), F(50 + 8), F(50 + 10), F(50 + 11),
                                  F(50 + 4), F(50 + 9), F(50 + 12), F(50 + 13),
                                  F(65), F(66), F(67), F(68), (float*)d_out);
}
